// GATv2modif_61263413510495
// MI455X (gfx1250) — matrix operand order repaired and checked by reading, not compiled, not run on hardware
//
#include <hip/hip_runtime.h>
#include <cstdint>
#include <cstddef>

// ---------------------------------------------------------------------------
// Problem constants (match reference)
// ---------------------------------------------------------------------------
#define NN      50000
#define EE      400000
#define IN_DIM  256
#define HEADS   4
#define CH      64
#define HC      256          // HEADS * CH
#define OUT_DIM 32
#define NEG_SLOPE 0.2f

typedef __attribute__((ext_vector_type(16))) _Float16 v16h;
typedef __attribute__((ext_vector_type(8)))  _Float16 h8;   // one b128 load
typedef __attribute__((ext_vector_type(4)))  _Float16 h4;   // one b64 store
typedef __attribute__((ext_vector_type(8)))  float    v8f;

// ---------------------------------------------------------------------------
// Helpers
// ---------------------------------------------------------------------------
__device__ __forceinline__ float lrelu(float v) {
    return v > 0.0f ? v : NEG_SLOPE * v;
}

// Monotone float->uint encoding so atomicMax(uint) == float max (any sign).
__device__ __forceinline__ uint32_t enc_f32(float f) {
    uint32_t b = __float_as_uint(f);
    return (b & 0x80000000u) ? ~b : (b | 0x80000000u);
}
__device__ __forceinline__ float dec_f32(uint32_t u) {
    uint32_t b = (u & 0x80000000u) ? (u ^ 0x80000000u) : ~u;
    return __uint_as_float(b);
}

__device__ __forceinline__ v16h cat8(h8 a, h8 b) {
    v16h r;
#pragma unroll
    for (int i = 0; i < 8; ++i) { r[i] = a[i]; r[i + 8] = b[i]; }
    return r;
}

// ---------------------------------------------------------------------------
// Pack pass: fp32 -> (hi, lo) f16 split, row-major, optional fused relu.
// a = hi + lo with |lo| <= ulp(hi)/2  ->  3-term WMMA product ~ fp32 accurate.
// ---------------------------------------------------------------------------
template <bool RELU>
__global__ void pack_a_kernel(const float* __restrict__ in,
                              _Float16* __restrict__ oh,
                              _Float16* __restrict__ ol,
                              int count4)           // count/4
{
    int t = blockIdx.x * blockDim.x + threadIdx.x;
    if (t >= count4) return;
    float4 v = ((const float4*)in)[t];
    if (RELU) {
        v.x = fmaxf(v.x, 0.f); v.y = fmaxf(v.y, 0.f);
        v.z = fmaxf(v.z, 0.f); v.w = fmaxf(v.w, 0.f);
    }
    h4 hi, lo;
    hi[0] = (_Float16)v.x; lo[0] = (_Float16)(v.x - (float)hi[0]);
    hi[1] = (_Float16)v.y; lo[1] = (_Float16)(v.y - (float)hi[1]);
    hi[2] = (_Float16)v.z; lo[2] = (_Float16)(v.z - (float)hi[2]);
    hi[3] = (_Float16)v.w; lo[3] = (_Float16)(v.w - (float)hi[3]);
    ((h4*)oh)[t] = hi;
    ((h4*)ol)[t] = lo;
}

// Pack weights transposed: W [K,Nn] fp32 -> Wt_hi/lo [Nn,K] f16.
__global__ void pack_bt_kernel(const float* __restrict__ W,
                               _Float16* __restrict__ oh,
                               _Float16* __restrict__ ol,
                               int K, int Nn)
{
    int t = blockIdx.x * blockDim.x + threadIdx.x;
    if (t >= K * Nn) return;
    int n = t / K, k = t - n * K;
    float v = W[(size_t)k * Nn + n];
    _Float16 hi = (_Float16)v;
    oh[t] = hi;
    ol[t] = (_Float16)(v - (float)hi);
}

// ---------------------------------------------------------------------------
// Packed split-f16 WMMA GEMM:  D = act(A @ B + bias)
//   AH/AL: row-major [M, K] f16 (hi/lo)
//   BH/BL: row-major [Nn, K] f16 (B transposed, hi/lo)
//   D:     row-major [M, Nn] fp32
// One wave computes a 16 x (16*NT) strip; inner loop = b128 loads + WMMAs.
// ACT: 0 = none, 2 = sigmoid
// ---------------------------------------------------------------------------
template <int NT, int ACT>
__global__ __launch_bounds__(32)
void wmma_gemm_packed(const _Float16* __restrict__ AH,
                      const _Float16* __restrict__ AL,
                      const _Float16* __restrict__ BH,
                      const _Float16* __restrict__ BL,
                      const float* __restrict__ bias,
                      float* __restrict__ D,
                      int Nn, int K)
{
    const int lane = threadIdx.x;               // 0..31, wave32
    const int m0   = blockIdx.x * 16;
    const int n0   = blockIdx.y * 16 * NT;

    // A fragment (16x32 f16, ISA 7.12.2): lane = {khalf, row}
    //   halves i<8 -> k = akb+i ; i>=8 -> k = akb+16+(i-8)  (two runs of 8)
    const int arow = lane & 15;
    const int akb  = (lane >> 4) * 8;           // 0 or 8
    // B fragment (32x16): lane = {khalf, col}; same k order as A (two runs of 8)
    const int bn   = lane & 15;
    const int bkb  = (lane >> 4) * 8;           // 0 or 8

    v8f acc[NT];
#pragma unroll
    for (int t = 0; t < NT; ++t)
#pragma unroll
        for (int r = 0; r < 8; ++r) acc[t][r] = 0.0f;

    const _Float16* arH = AH + (size_t)(m0 + arow) * K + akb;
    const _Float16* arL = AL + (size_t)(m0 + arow) * K + akb;

    for (int k0 = 0; k0 < K; k0 += 32) {
        v16h a_hi = cat8(*(const h8*)(arH + k0), *(const h8*)(arH + k0 + 16));
        v16h a_lo = cat8(*(const h8*)(arL + k0), *(const h8*)(arL + k0 + 16));

#pragma unroll
        for (int t = 0; t < NT; ++t) {
            const size_t boff = (size_t)(n0 + t * 16 + bn) * K + k0 + bkb;
            v16h b_hi = cat8(*(const h8*)(BH + boff), *(const h8*)(BH + boff + 16));
            v16h b_lo = cat8(*(const h8*)(BL + boff), *(const h8*)(BL + boff + 16));

            acc[t] = __builtin_amdgcn_wmma_f32_16x16x32_f16(
                         false, a_hi, false, b_hi, (short)0, acc[t], false, false);
            acc[t] = __builtin_amdgcn_wmma_f32_16x16x32_f16(
                         false, a_hi, false, b_lo, (short)0, acc[t], false, false);
            acc[t] = __builtin_amdgcn_wmma_f32_16x16x32_f16(
                         false, a_lo, false, b_hi, (short)0, acc[t], false, false);
        }
    }

    // Epilogue; C layout: M = 8*(lane>>4)+r, N = lane&15
    const int mo = (lane >> 4) * 8;
#pragma unroll
    for (int t = 0; t < NT; ++t) {
        const int n  = n0 + t * 16 + (lane & 15);
        const float bv = bias[n];
#pragma unroll
        for (int r = 0; r < 8; ++r) {
            float v = acc[t][r] + bv;
            if (ACT == 2) v = 1.0f / (1.0f + __expf(-v));
            D[(size_t)(m0 + mo + r) * Nn + n] = v;
        }
    }
}

// ---------------------------------------------------------------------------
// Edge kernels (GATv2 attention, segment softmax by dst) — L2-resident
// ---------------------------------------------------------------------------
__global__ void edge_logits_kernel(const float* __restrict__ Hfeat,
                                   const long long* __restrict__ esrc,
                                   const long long* __restrict__ edst,
                                   const float* __restrict__ att,
                                   float* __restrict__ logit,
                                   uint32_t* __restrict__ amax)
{
    int t = blockIdx.x * blockDim.x + threadIdx.x;
    if (t >= EE * HEADS) return;
    const int e = t >> 2;
    const int h = t & 3;
    const int s = (int)esrc[e];
    const int d = (int)edst[e];

    const float4* xs = (const float4*)(Hfeat + (size_t)s * HC + h * CH);
    const float4* xd = (const float4*)(Hfeat + (size_t)d * HC + h * CH);
    const float4* at = (const float4*)(att + h * CH);

    float sum = 0.0f;
#pragma unroll
    for (int i = 0; i < CH / 4; ++i) {
        float4 a = xs[i], b = xd[i], w = at[i];
        sum += w.x * lrelu(a.x + b.x);
        sum += w.y * lrelu(a.y + b.y);
        sum += w.z * lrelu(a.z + b.z);
        sum += w.w * lrelu(a.w + b.w);
    }
    logit[t] = sum;
    atomicMax(amax + (size_t)d * HEADS + h, enc_f32(sum));
}

__global__ void edge_exp_kernel(const long long* __restrict__ edst,
                                const float* __restrict__ logit,
                                const uint32_t* __restrict__ amax,
                                float* __restrict__ exs,
                                float* __restrict__ den)
{
    int t = blockIdx.x * blockDim.x + threadIdx.x;
    if (t >= EE * HEADS) return;
    const int e = t >> 2;
    const int h = t & 3;
    const int d = (int)edst[e];
    float m  = dec_f32(amax[(size_t)d * HEADS + h]);
    float ex = __expf(logit[t] - m);
    exs[t] = ex;
    atomicAdd(den + (size_t)d * HEADS + h, ex);
}

__global__ void edge_scatter_kernel(const float* __restrict__ Hfeat,
                                    const long long* __restrict__ esrc,
                                    const long long* __restrict__ edst,
                                    const float* __restrict__ exs,
                                    const float* __restrict__ den,
                                    float* __restrict__ agg)
{
    int t = blockIdx.x * blockDim.x + threadIdx.x;
    if (t >= EE * (HC / 4)) return;
    const int e = t >> 6;                 // 64 4-wide chunks per edge
    const int q = t & 63;
    const int h = q >> 4;
    const int s = (int)esrc[e];
    const int d = (int)edst[e];

    const float w = exs[(size_t)e * HEADS + h] /
                    (den[(size_t)d * HEADS + h] + 1e-16f);
    const float4 xv = *(const float4*)(Hfeat + (size_t)s * HC + q * 4);
    float* out = agg + (size_t)d * HC + q * 4;
    atomicAdd(out + 0, xv.x * w);
    atomicAdd(out + 1, xv.y * w);
    atomicAdd(out + 2, xv.z * w);
    atomicAdd(out + 3, xv.w * w);
}

// ---------------------------------------------------------------------------
// Host-side launch
// ---------------------------------------------------------------------------
extern "C" void kernel_launch(void* const* d_in, const int* in_sizes, int n_in,
                              void* d_out, int out_size, void* d_ws, size_t ws_size,
                              hipStream_t stream)
{
    (void)in_sizes; (void)n_in; (void)out_size; (void)ws_size;

    const float*     x    = (const float*)d_in[0];
    const long long* ei   = (const long long*)d_in[1];
    const float*     W1   = (const float*)d_in[2];
    const float*     b1   = (const float*)d_in[3];
    const float*     att1 = (const float*)d_in[4];
    const float*     W2   = (const float*)d_in[5];
    const float*     b2   = (const float*)d_in[6];
    const float*     att2 = (const float*)d_in[7];
    const float*     Wp1  = (const float*)d_in[8];
    const float*     bp1  = (const float*)d_in[9];
    const float*     Wp2  = (const float*)d_in[10];
    const float*     bp2  = (const float*)d_in[11];

    const long long* esrc = ei;        // edge_index[0, :]
    const long long* edst = ei + EE;   // edge_index[1, :]

    // ---- workspace layout ----
    auto align256 = [](size_t v) { return (v + 255) & ~(size_t)255; };
    char* ws = (char*)d_ws;
    const size_t SZ_H     = align256((size_t)NN * HC * 4);       // 51.2 MB
    const size_t SZ_AGG   = align256((size_t)NN * HC * 4);       // 51.2 MB
    const size_t SZ_LOGIT = align256((size_t)EE * HEADS * 4);
    const size_t SZ_EXS   = align256((size_t)EE * HEADS * 4);
    const size_t SZ_AMAX  = align256((size_t)NN * HEADS * 4);
    const size_t SZ_DEN   = align256((size_t)NN * HEADS * 4);
    const size_t SZ_T     = align256((size_t)NN * CH * 4);       // 12.8 MB
    const size_t SZ_APK   = align256((size_t)NN * HC * 2);       // 25.6 MB each

    float*     H     = (float*)ws;      ws += SZ_H;
    float*     AGG   = (float*)ws;      ws += SZ_AGG;
    float*     LOGIT = (float*)ws;      ws += SZ_LOGIT;
    float*     EXS   = (float*)ws;      ws += SZ_EXS;
    uint32_t*  AMAX  = (uint32_t*)ws;   ws += SZ_AMAX;
    float*     DEN   = (float*)ws;      ws += SZ_DEN;
    float*     T     = (float*)ws;      ws += SZ_T;
    _Float16*  AHp   = (_Float16*)ws;   ws += SZ_APK;
    _Float16*  ALp   = (_Float16*)ws;   ws += SZ_APK;
    // packed transposed weights (hi/lo), tiny
    _Float16* W1H = (_Float16*)ws;  ws += align256((size_t)HC * IN_DIM * 2);
    _Float16* W1L = (_Float16*)ws;  ws += align256((size_t)HC * IN_DIM * 2);
    _Float16* W2H = (_Float16*)ws;  ws += align256((size_t)HC * HC * 2);
    _Float16* W2L = (_Float16*)ws;  ws += align256((size_t)HC * HC * 2);
    _Float16* P1H = (_Float16*)ws;  ws += align256((size_t)CH * HC * 2);
    _Float16* P1L = (_Float16*)ws;  ws += align256((size_t)CH * HC * 2);
    _Float16* P2H = (_Float16*)ws;  ws += align256((size_t)OUT_DIM * CH * 2);
    _Float16* P2L = (_Float16*)ws;

    const dim3 blk32(32);
    const int  TB = 256;
    const dim3 gemm_grid_256(NN / 16, HC / 64);   // NT=4
    const dim3 gemm_grid_1  (NN / 16, 1);
    const dim3 egrid_eh((EE * HEADS + TB - 1) / TB);
    const dim3 egrid_sc((EE * (HC / 4) + TB - 1) / TB);
    auto g1d = [&](size_t n) { return dim3((unsigned)((n + TB - 1) / TB)); };

    // ---- pack all weights (transposed, hi/lo) ----
    pack_bt_kernel<<<g1d((size_t)IN_DIM * HC), TB, 0, stream>>>(W1, W1H, W1L, IN_DIM, HC);
    pack_bt_kernel<<<g1d((size_t)HC * HC),     TB, 0, stream>>>(W2, W2H, W2L, HC, HC);
    pack_bt_kernel<<<g1d((size_t)HC * CH),     TB, 0, stream>>>(Wp1, P1H, P1L, HC, CH);
    pack_bt_kernel<<<g1d((size_t)CH * OUT_DIM),TB, 0, stream>>>(Wp2, P2H, P2L, CH, OUT_DIM);

    // ================= layer 1 =================
    pack_a_kernel<false><<<g1d((size_t)NN * IN_DIM / 4), TB, 0, stream>>>(
        x, AHp, ALp, NN * IN_DIM / 4);
    wmma_gemm_packed<4, 0><<<gemm_grid_256, blk32, 0, stream>>>(
        AHp, ALp, W1H, W1L, b1, H, HC, IN_DIM);

    hipMemsetAsync(AMAX, 0, (size_t)NN * HEADS * 4, stream);  // 0 == -inf code
    hipMemsetAsync(DEN,  0, (size_t)NN * HEADS * 4, stream);
    hipMemsetAsync(AGG,  0, (size_t)NN * HC * 4, stream);

    edge_logits_kernel<<<egrid_eh, TB, 0, stream>>>(H, esrc, edst, att1, LOGIT, AMAX);
    edge_exp_kernel<<<egrid_eh, TB, 0, stream>>>(edst, LOGIT, AMAX, EXS, DEN);
    edge_scatter_kernel<<<egrid_sc, TB, 0, stream>>>(H, esrc, edst, EXS, DEN, AGG);

    // ================= layer 2 =================
    pack_a_kernel<true><<<g1d((size_t)NN * HC / 4), TB, 0, stream>>>(
        AGG, AHp, ALp, NN * HC / 4);                 // relu fused
    wmma_gemm_packed<4, 0><<<gemm_grid_256, blk32, 0, stream>>>(
        AHp, ALp, W2H, W2L, b2, H, HC, HC);

    hipMemsetAsync(AMAX, 0, (size_t)NN * HEADS * 4, stream);
    hipMemsetAsync(DEN,  0, (size_t)NN * HEADS * 4, stream);
    hipMemsetAsync(AGG,  0, (size_t)NN * HC * 4, stream);

    edge_logits_kernel<<<egrid_eh, TB, 0, stream>>>(H, esrc, edst, att2, LOGIT, AMAX);
    edge_exp_kernel<<<egrid_eh, TB, 0, stream>>>(edst, LOGIT, AMAX, EXS, DEN);
    edge_scatter_kernel<<<egrid_sc, TB, 0, stream>>>(H, esrc, edst, EXS, DEN, AGG);

    // ================= post_mp =================
    pack_a_kernel<true><<<g1d((size_t)NN * HC / 4), TB, 0, stream>>>(
        AGG, AHp, ALp, NN * HC / 4);                 // relu fused
    wmma_gemm_packed<4, 0><<<gemm_grid_1, blk32, 0, stream>>>(
        AHp, ALp, P1H, P1L, bp1, T, CH, HC);         // T: [NN,64]

    pack_a_kernel<false><<<g1d((size_t)NN * CH / 4), TB, 0, stream>>>(
        T, AHp, ALp, NN * CH / 4);
    wmma_gemm_packed<2, 2><<<gemm_grid_1, blk32, 0, stream>>>(
        AHp, ALp, P2H, P2L, bp2, (float*)d_out, OUT_DIM, CH);  // sigmoid
}
